// Seq2SeqLSTM_71906342470204
// MI455X (gfx1250) — matrix operand order repaired and checked by reading, not compiled, not run on hardware
//
#include <hip/hip_runtime.h>

typedef __attribute__((ext_vector_type(16))) __bf16 v16bf;
typedef __attribute__((ext_vector_type(8)))  float  v8f;

#define NBLK 16

// ---------------- workspace layout (bytes) ----------------
constexpr size_t OFF_ENCW = 0;                                   // 16 cells * 4 MB bf16 B-images
constexpr size_t OFF_DECW = OFF_ENCW + (size_t)16 * 4194304;     // 67,108,864
constexpr size_t OFF_XIMG = OFF_DECW + (size_t)16 * 4194304;     // 134,217,728 (256 steps * 64KB A-images)
constexpr size_t OFF_FINW = OFF_XIMG + (size_t)256 * 65536;      // 150,994,944 (512KB)
constexpr size_t OFF_YSA  = OFF_FINW + (size_t)524288;           // 151,519,232 (4096 mtiles * 16KB)
constexpr size_t OFF_ENCH = OFF_YSA + (size_t)4096 * 16384;      // 218,628,096 (4 layers * 2 bufs * 64KB)
constexpr size_t OFF_DECH = OFF_ENCH + (size_t)8 * 65536;        // 219,152,384
constexpr size_t OFF_BAR  = OFF_DECH + (size_t)8 * 65536;        // 219,676,672

// ---------------- helpers ----------------
__device__ __forceinline__ unsigned short f2bf(float x) {
  unsigned u = __float_as_uint(x);
  u += 0x7FFFu + ((u >> 16) & 1u);           // round-to-nearest-even
  return (unsigned short)(u >> 16);
}
__device__ __forceinline__ float sigmoidf_(float x) { return 1.0f / (1.0f + __expf(-x)); }
__device__ __forceinline__ float tanhf_(float x)    { return 1.0f - 2.0f / (1.0f + __expf(2.0f * x)); }

// sense-reversing device-wide barrier (bar[0]=arrive, bar[1]=generation)
__device__ __forceinline__ void grid_barrier(unsigned* bar, int nblk) {
  __threadfence();
  __syncthreads();
  if (threadIdx.x == 0) {
    unsigned gen  = __hip_atomic_load(&bar[1], __ATOMIC_RELAXED, __HIP_MEMORY_SCOPE_AGENT);
    unsigned prev = __hip_atomic_fetch_add(&bar[0], 1u, __ATOMIC_ACQ_REL, __HIP_MEMORY_SCOPE_AGENT);
    if (prev == (unsigned)(nblk - 1)) {
      __hip_atomic_store(&bar[0], 0u, __ATOMIC_RELAXED, __HIP_MEMORY_SCOPE_AGENT);
      __hip_atomic_store(&bar[1], gen + 1u, __ATOMIC_RELEASE, __HIP_MEMORY_SCOPE_AGENT);
    } else {
      while (__hip_atomic_load(&bar[1], __ATOMIC_ACQUIRE, __HIP_MEMORY_SCOPE_AGENT) == gen)
        __builtin_amdgcn_s_sleep(2);
    }
  }
  __syncthreads();
}

// ---------------- prep: pack weights into WMMA B-operand images ----------------
// image ushort index = cell*2097152 + gnt*16384 + ks*512 + lane*16 + slot
// B 32x16 bf16 layout: kk = (slot&7) + 8*(lane>>4) + 16*(slot>>3), N=lane&15 (matches A image).
// gate-interleaved column permutation: gnt = wg*8 + tIdx, gate=tIdx>>1, sub=tIdx&1,
//   col = gate*512 + wg*32 + sub*16 + nIn ; ks<16 -> W (K 0..511), ks>=16 -> V.
__global__ void prep_weights(const float* __restrict__ W, const float* __restrict__ V,
                             unsigned short* __restrict__ img) {
  size_t idx = (size_t)blockIdx.x * 256 + threadIdx.x;   // 33,554,432 total
  int slot = (int)(idx & 15);
  int lane = (int)((idx >> 4) & 31);
  int ks   = (int)((idx >> 9) & 31);
  int gnt  = (int)((idx >> 14) & 127);
  int cell = (int)(idx >> 21);
  int nIn = lane & 15;
  int kk  = (slot & 7) + ((lane >> 4) << 3) + ((slot >> 3) << 4);
  int wg = gnt >> 3, tIdx = gnt & 7;
  int gate = tIdx >> 1, sub = tIdx & 1;
  int col = gate * 512 + wg * 32 + sub * 16 + nIn;
  int k = (ks & 15) * 32 + kk;
  const float* src = (ks < 16) ? W : V;
  img[idx] = f2bf(src[(size_t)cell * (512 * 2048) + (size_t)k * 2048 + col]);
}

// A 16x32 bf16 operand-image mapping: lane = (m&15) + 16*((kk>>3)&1),
// slot = (kk&7) + 8*(kk>>4)  (inverse used below: kk = (slot&7) + 8*(lane>>4) + 16*(slot>>3))
__global__ void prep_x(const float* __restrict__ x, unsigned short* __restrict__ img) {
  size_t idx = (size_t)blockIdx.x * 256 + threadIdx.x;   // 8,388,608 total
  int slot = (int)(idx & 15);
  int lane = (int)((idx >> 4) & 31);
  int ks   = (int)((idx >> 9) & 15);
  int mt   = (int)((idx >> 13) & 3);
  int t    = (int)(idx >> 15);
  int m  = mt * 16 + (lane & 15);
  int kk = (slot & 7) + ((lane >> 4) << 3) + ((slot >> 3) << 4);
  int k  = ks * 32 + kk;
  img[idx] = f2bf(x[((size_t)m * 256 + t) * 512 + k]);   // x is (64,256,512)
}

__global__ void prep_h(const float* __restrict__ encH0, const float* __restrict__ decH0,
                       unsigned short* __restrict__ encImg, unsigned short* __restrict__ decImg) {
  size_t idx = (size_t)blockIdx.x * 256 + threadIdx.x;   // 262,144 total
  int slot = (int)(idx & 15);
  int lane = (int)((idx >> 4) & 31);
  int ks   = (int)((idx >> 9) & 15);
  int mt   = (int)((idx >> 13) & 3);
  int e    = (int)((idx >> 15) & 3);
  int which = (int)(idx >> 17);
  int m  = mt * 16 + (lane & 15);
  int kk = (slot & 7) + ((lane >> 4) << 3) + ((slot >> 3) << 4);
  int k  = ks * 32 + kk;
  const float* src = which ? decH0 : encH0;
  unsigned short* dst = which ? decImg : encImg;
  // initial state goes into parity buffer 1
  dst[(size_t)(e * 2 + 1) * 32768 + (size_t)mt * 8192 + (size_t)ks * 512 + lane * 16 + slot]
      = f2bf(src[((size_t)e * 64 + m) * 512 + k]);
}

__global__ void prep_finw(const float* __restrict__ fw, unsigned short* __restrict__ img,
                          unsigned* __restrict__ bar) {
  size_t idx = (size_t)blockIdx.x * 256 + threadIdx.x;   // 262,144 total
  if (idx < 2) bar[idx] = 0;                             // reset grid barrier each launch
  int slot = (int)(idx & 15);
  int lane = (int)((idx >> 4) & 31);
  int ks   = (int)((idx >> 9) & 15);
  int nt   = (int)(idx >> 13);
  int nIn = lane & 15;
  int kk  = (slot & 7) + ((lane >> 4) << 3) + ((slot >> 3) << 4);
  int k   = ks * 32 + kk;
  int col = nt * 16 + nIn;
  img[idx] = f2bf(fw[(size_t)k * 512 + col]);            // fin_W is (512,512)
}

// ---------------- persistent recurrent phase (encoder DEC=0, decoder DEC=1) ----------------
template <int DEC>
__global__ __launch_bounds__(1024, 1) void lstm_phase(
    const unsigned short* __restrict__ ximg,    // encoder input images (enc only)
    const unsigned short* __restrict__ Wimg,    // packed W|V B-images, 16 cells
    const float* __restrict__ bias,             // (4,4,2048) fp32
    unsigned short* __restrict__ himg,          // 4 layers x 2 parity bufs x 64KB A-images
    const float* __restrict__ c0,               // (4,64,512) fp32
    const unsigned short* __restrict__ state0,  // dec only: encoder final h[3] image
    unsigned short* __restrict__ ysA,           // dec only: output A-images for final GEMM
    unsigned* __restrict__ bar) {
  __shared__ float ldsZ[8][4][16][16];   // 32 KB: per-wave D tiles (gate exchange)
  __shared__ float ldsC[4][64][32];      // 32 KB: persistent cell state c for this wg's 32 units
  const int wg   = blockIdx.x;
  const int tid  = threadIdx.x;
  const int lane = tid & 31;
  const int wave = tid >> 5;
  const int tIdx = wave & 7;             // ntile within wg: 0-1 f, 2-3 i, 4-5 g, 6-7 o
  const int mt   = wave >> 3;            // mtile 0..3
  const int gnt  = wg * 8 + tIdx;

  for (int u = tid; u < 4 * 64 * 32; u += 1024) {
    int e = u >> 11, b = (u >> 5) & 63, jj = u & 31;
    ldsC[e][b][jj] = c0[((size_t)e * 64 + b) * 512 + wg * 32 + jj];
  }
  __syncthreads();

  for (int t = 0; t < 256; ++t) {
    for (int L = 0; L < 4; ++L) {
      for (int e = 0; e < 4; ++e) {
        const int s = t * 4 + L;         // write-parity counter for this cell group
        const unsigned short* A1;
        if (e == 0) {
          if (DEC) A1 = (s == 0) ? state0 : himg + (size_t)(6 + ((s - 1) & 1)) * 32768;
          else     A1 = ximg + (size_t)t * 32768;
        } else {
          A1 = himg + (size_t)((e - 1) * 2 + (s & 1)) * 32768;
        }
        const unsigned short* A2 = himg + (size_t)(e * 2 + ((s - 1) & 1)) * 32768;
        const int cell = L * 4 + e;
        const v16bf* ap1 = (const v16bf*)(A1 + (size_t)mt * 8192 + lane * 16);
        const v16bf* ap2 = (const v16bf*)(A2 + (size_t)mt * 8192 + lane * 16);
        const v16bf* bp  = (const v16bf*)(Wimg + (size_t)cell * 2097152 + (size_t)gnt * 16384 + lane * 16);

        v8f acc = {0.f, 0.f, 0.f, 0.f, 0.f, 0.f, 0.f, 0.f};
#pragma unroll
        for (int ks = 0; ks < 16; ++ks)   // inp @ W  (K = 0..511)
          acc = __builtin_amdgcn_wmma_f32_16x16x32_bf16(false, ap1[ks * 32], false, bp[ks * 32],
                                                        (short)0, acc, false, false);
#pragma unroll
        for (int ks = 0; ks < 16; ++ks)   // h @ V    (K = 512..1023)
          acc = __builtin_amdgcn_wmma_f32_16x16x32_bf16(false, ap2[ks * 32], false, bp[(16 + ks) * 32],
                                                        (short)0, acc, false, false);

        // D tile -> LDS (C/D layout: vgpr r, lane l -> m = r + 8*(l>>4), n = l&15)
#pragma unroll
        for (int r = 0; r < 8; ++r)
          ldsZ[tIdx][mt][r + ((lane >> 4) << 3)][lane & 15] = acc[r];
        __syncthreads();

        unsigned short* hout = himg + (size_t)(e * 2 + (s & 1)) * 32768;
        const float* bcell = bias + (size_t)cell * 2048;
        for (int u = tid; u < 2048; u += 1024) {
          int b = u >> 5, jj = u & 31;
          int mtl = b >> 4, mi = b & 15;
          int nsub = jj >> 4, ni = jj & 15;
          int j = wg * 32 + jj;
          float zf = ldsZ[0 + nsub][mtl][mi][ni] + bcell[j];
          float zi = ldsZ[2 + nsub][mtl][mi][ni] + bcell[512 + j];
          float zg = ldsZ[4 + nsub][mtl][mi][ni] + bcell[1024 + j];
          float zo = ldsZ[6 + nsub][mtl][mi][ni] + bcell[1536 + j];
          float fg = sigmoidf_(zf);
          float ig = sigmoidf_(zi);
          float gg = tanhf_(zg);
          float og = sigmoidf_(zo);
          float C = fg * ldsC[e][b][jj] + ig * gg;
          float H = og * tanhf_(C);
          ldsC[e][b][jj] = C;
          int ks = j >> 5, kk = j & 31;
          int ln = mi + (((kk >> 3) & 1) << 4);
          int sl = (kk & 7) + ((kk >> 4) << 3);
          hout[(size_t)(mtl * 16 + ks) * 512 + ln * 16 + sl] = f2bf(H);
          if (DEC && e == 3) {            // append state to ys A-image (row r = s)
            int m2t = b * 64 + (s >> 4);
            int ln2 = (s & 15) + (((kk >> 3) & 1) << 4);
            ysA[(size_t)(m2t * 16 + ks) * 512 + ln2 * 16 + sl] = f2bf(H);
          }
        }
        grid_barrier(bar, NBLK);          // one device-wide barrier per cell
      }
    }
  }
}

// ---------------- final projection: sigmoid((65536x512) @ (512x512) + b) ----------------
__global__ __launch_bounds__(1024, 1) void fin_kernel(
    const unsigned short* __restrict__ ysA, const unsigned short* __restrict__ Fimg,
    const float* __restrict__ fb, float* __restrict__ out) {
  const int lane = threadIdx.x & 31;
  const int wave = threadIdx.x >> 5;
  const int tile = blockIdx.x * 32 + wave;
  const int nt = tile & 31;
  const int mtile = tile >> 5;
  const v16bf* ap = (const v16bf*)(ysA + (size_t)mtile * 8192 + lane * 16);
  const v16bf* bp = (const v16bf*)(Fimg + (size_t)nt * 8192 + lane * 16);
  v8f acc = {0.f, 0.f, 0.f, 0.f, 0.f, 0.f, 0.f, 0.f};
#pragma unroll
  for (int ks = 0; ks < 16; ++ks)
    acc = __builtin_amdgcn_wmma_f32_16x16x32_bf16(false, ap[ks * 32], false, bp[ks * 32],
                                                  (short)0, acc, false, false);
  int n = nt * 16 + (lane & 15);
  float bb = fb[n];
  int mbase = mtile * 16 + ((lane >> 4) << 3);
#pragma unroll
  for (int r = 0; r < 8; ++r)
    out[(size_t)(mbase + r) * 512 + n] = sigmoidf_(acc[r] + bb);
}

// ---------------- launcher ----------------
extern "C" void kernel_launch(void* const* d_in, const int* in_sizes, int n_in,
                              void* d_out, int out_size, void* d_ws, size_t ws_size,
                              hipStream_t stream) {
  (void)in_sizes; (void)n_in; (void)out_size; (void)ws_size;
  const float* x     = (const float*)d_in[0];
  const float* encW  = (const float*)d_in[2];
  const float* encV  = (const float*)d_in[3];
  const float* encB  = (const float*)d_in[4];
  const float* encH0 = (const float*)d_in[5];
  const float* encC0 = (const float*)d_in[6];
  const float* decW  = (const float*)d_in[7];
  const float* decV  = (const float*)d_in[8];
  const float* decB  = (const float*)d_in[9];
  const float* decH0 = (const float*)d_in[10];
  const float* decC0 = (const float*)d_in[11];
  const float* finW  = (const float*)d_in[12];
  const float* finB  = (const float*)d_in[13];

  char* ws = (char*)d_ws;
  unsigned short* encWimg = (unsigned short*)(ws + OFF_ENCW);
  unsigned short* decWimg = (unsigned short*)(ws + OFF_DECW);
  unsigned short* ximg    = (unsigned short*)(ws + OFF_XIMG);
  unsigned short* finWimg = (unsigned short*)(ws + OFF_FINW);
  unsigned short* ysA     = (unsigned short*)(ws + OFF_YSA);
  unsigned short* encHimg = (unsigned short*)(ws + OFF_ENCH);
  unsigned short* decHimg = (unsigned short*)(ws + OFF_DECH);
  unsigned*       bar     = (unsigned*)(ws + OFF_BAR);

  prep_weights<<<131072, 256, 0, stream>>>(encW, encV, encWimg);
  prep_weights<<<131072, 256, 0, stream>>>(decW, decV, decWimg);
  prep_x<<<32768, 256, 0, stream>>>(x, ximg);
  prep_h<<<1024, 256, 0, stream>>>(encH0, decH0, encHimg, decHimg);
  prep_finw<<<1024, 256, 0, stream>>>(finW, finWimg, bar);

  lstm_phase<0><<<NBLK, 1024, 0, stream>>>(ximg, encWimg, encB, encHimg, encC0,
                                           nullptr, nullptr, bar);
  // encoder layer-3 final value lives in parity buffer 1
  lstm_phase<1><<<NBLK, 1024, 0, stream>>>(nullptr, decWimg, decB, decHimg, decC0,
                                           encHimg + (size_t)(3 * 2 + 1) * 32768, ysA, bar);

  fin_kernel<<<4096, 1024, 0, stream>>>(ysA, finWimg, finB, (float*)d_out);
}
